// Fingerprint_70145405878925
// MI455X (gfx1250) — matrix operand order repaired and checked by reading, not compiled, not run on hardware
//
#include <hip/hip_runtime.h>
#include <hip/hip_bf16.h>
#include <cstdint>
#include <cstddef>

// ---------------------------------------------------------------------------
// AttentiveFP forward for MI455X (gfx1250, wave32, WMMA bf16 16x16x32).
// All heavy GEMMs run through v_wmma_f32_16x16x32_bf16; attention weighting
// is folded in front of the attend GEMMs (linearity), shrinking the attend
// step from [98304x256]x[256x256] to [16384x256]x[256x256].
// ---------------------------------------------------------------------------

typedef __bf16 bf16_t;
typedef __attribute__((ext_vector_type(16))) __bf16 v16bf;
typedef __attribute__((ext_vector_type(8)))  __bf16 v8bf;
typedef __attribute__((ext_vector_type(8)))  float  v8f;

#define NB   256      // batch
#define NL   64       // atoms per molecule
#define NDEG 6
#define NFP  256
#define NFEAT 39
#define NBOND 10
#define NEGC (-9.0e8f)

__device__ __forceinline__ float leakyf_(float x)  { return x >= 0.f ? x : 0.01f * x; }
__device__ __forceinline__ float sigmoidf_(float x){ return 1.f / (1.f + __expf(-x)); }
__device__ __forceinline__ float eluf_(float x)    { return x > 0.f ? x : (__expf(x) - 1.f); }

// ---------------------------------------------------------------------------
// Generic bf16 GEMM:  C[M,N] = epi( A[M,K] @ W[N,K]^T + rowscale[m]*bias[n] )
// One wave owns a 16x64 strip of C (4 WMMA tiles sharing one A fragment).
// EPI: 0 = none, 1 = leaky_relu(0.01), 2 = ELU.
// ---------------------------------------------------------------------------
template <int EPI>
__global__ __launch_bounds__(256) void gemm_bf16_kernel(
    const bf16_t* __restrict__ A, const bf16_t* __restrict__ W,
    const float* __restrict__ bias, const float* __restrict__ rowscale,
    float* __restrict__ Cf, bf16_t* __restrict__ Cb,
    int M, int N, int K)
{
  const int wave = threadIdx.x >> 5;
  const int lane = threadIdx.x & 31;
  const int half = lane >> 4;      // K-half selector (A/B frag layout)
  const int lr   = lane & 15;      // row (A) / col (B) within tile
  const int mTile = blockIdx.x * 8 + wave;
  if (mTile * 16 >= M) return;
  const int nBase = blockIdx.y * 64;
  const int arow  = mTile * 16 + lr;

  v8f acc[4] = {};
  for (int k0 = 0; k0 < K; k0 += 32) {
    // A frag (16x32 bf16): lane lr holds row M=lr; 16 contiguous K values
    // starting at k0 + half*16.
    const bf16_t* ap = A + (size_t)arow * K + k0 + half * 16;
    v8bf a0 = *reinterpret_cast<const v8bf*>(ap);
    v8bf a1 = *reinterpret_cast<const v8bf*>(ap + 8);
    v16bf afrag = __builtin_shufflevector(a0, a1,
        0, 1, 2, 3, 4, 5, 6, 7, 8, 9, 10, 11, 12, 13, 14, 15);
#pragma unroll
    for (int t = 0; t < 4; ++t) {
      // B frag (32x16 bf16): lane lr holds column N; 16 contiguous K values
      // starting at k0 + half*16.  W is row-major [N,K] (x @ W^T layout).
      const bf16_t* wp = W + (size_t)(nBase + t * 16 + lr) * K + k0 + half * 16;
      v16bf bfrag = *reinterpret_cast<const v16bf*>(wp);
      acc[t] = __builtin_amdgcn_wmma_f32_16x16x32_bf16(
          false, afrag, false, bfrag, (short)0, acc[t], false, false);
    }
  }

  // C layout: lanes 0-15 -> M = r, lanes 16-31 -> M = 8 + r (VGPR index r).
  const int orow0 = mTile * 16 + half * 8;
#pragma unroll
  for (int t = 0; t < 4; ++t) {
    const int col = nBase + t * 16 + lr;
#pragma unroll
    for (int r = 0; r < 8; ++r) {
      const int orow = orow0 + r;
      float v = acc[t][r];
      if (bias) {
        float b = bias[col];
        if (rowscale) b *= rowscale[orow];
        v += b;
      }
      if (EPI == 1) v = leakyf_(v);
      else if (EPI == 2) v = eluf_(v);
      const size_t idx = (size_t)orow * N + col;
      if (Cf) Cf[idx] = v;
      if (Cb) Cb[idx] = (bf16_t)v;
    }
  }
}

// fp32 -> bf16 with K-padding: dst is [N, Kp], src is [N, K], pad zeros.
__global__ void convert_pad_kernel(const float* __restrict__ src,
                                   bf16_t* __restrict__ dst,
                                   int K, int Kp, size_t total)
{
  size_t i = (size_t)blockIdx.x * blockDim.x + threadIdx.x;
  if (i >= total) return;
  size_t n = i / (size_t)Kp;
  int k = (int)(i - n * (size_t)Kp);
  dst[i] = (k < K) ? (bf16_t)src[n * (size_t)K + k] : (bf16_t)0.f;
}

// Build gathered neighbor input [B*L*DEG, 64]: [atom(39) | bond(10) | pad].
__global__ void build_neighbor_kernel(const float* __restrict__ atom_list,
                                      const float* __restrict__ bond_list,
                                      const int* __restrict__ adeg,
                                      const int* __restrict__ bdeg,
                                      bf16_t* __restrict__ out)
{
  size_t tid = (size_t)blockIdx.x * blockDim.x + threadIdx.x;
  size_t m = tid >> 6;            // (b, l, d) flat
  int k = (int)(tid & 63);
  if (m >= (size_t)NB * NL * NDEG) return;
  int b = (int)(m / (NL * NDEG));
  float v = 0.f;
  if (k < NFEAT) {
    int aj = adeg[m];
    v = atom_list[((size_t)b * NL + aj) * NFEAT + k];
  } else if (k < NFEAT + NBOND) {
    int bj = bdeg[m];
    v = bond_list[((size_t)b * NL + bj) * NBOND + (k - NFEAT)];
  }
  out[tid] = (bf16_t)v;
}

// Per-(b,l): rank-1 align scores over DEG neighbors, masked softmax, then
// weighted neighbor-feature sum (folds attention in front of attend GEMM).
// selfmode: neighbor feature = relu(atom_feature) (radius >= 1).
__global__ __launch_bounds__(256) void align_attend_kernel(
    const float* __restrict__ af, const bf16_t* __restrict__ nf,
    const int* __restrict__ adeg,
    const float* __restrict__ aw, const float* __restrict__ ab_p,
    bf16_t* __restrict__ wsum_bf, float* __restrict__ swout, int selfmode)
{
  const int bl = blockIdx.x;        // b*L + l
  const int t  = threadIdx.x;       // feature index 0..255
  __shared__ float sred[256];
  __shared__ float res[7];
  __shared__ float wsh[NDEG];

  float afv = af[(size_t)bl * NFP + t];
  float selfv = fmaxf(afv, 0.f);
  float nfv[NDEG];
#pragma unroll
  for (int d = 0; d < NDEG; ++d) {
    if (selfmode) nfv[d] = selfv;
    else          nfv[d] = (float)nf[((size_t)bl * NDEG + d) * NFP + t];
  }
  float aw_a = aw[t], aw_b = aw[NFP + t];
  float part[7];
  part[0] = afv * aw_a;
#pragma unroll
  for (int d = 0; d < NDEG; ++d) part[1 + d] = nfv[d] * aw_b;

  for (int i = 0; i < 7; ++i) {
    sred[t] = part[i];
    __syncthreads();
    for (int s = 128; s > 0; s >>= 1) {
      if (t < s) sred[t] += sred[t + s];
      __syncthreads();
    }
    if (t == 0) res[i] = sred[0];
    __syncthreads();
  }

  if (t == 0) {
    float ab = ab_p[0];
    float sc[NDEG];
    float mx = -3.4e38f;
#pragma unroll
    for (int d = 0; d < NDEG; ++d) {
      float s = leakyf_(res[0] + res[1 + d] + ab);
      if (adeg[bl * NDEG + d] == NL - 1) s += NEGC;  // softmax mask
      sc[d] = s;
      mx = fmaxf(mx, s);
    }
    float den = 0.f;
#pragma unroll
    for (int d = 0; d < NDEG; ++d) { sc[d] = __expf(sc[d] - mx); den += sc[d]; }
    float sw = 0.f;
#pragma unroll
    for (int d = 0; d < NDEG; ++d) {
      float w = (adeg[bl * NDEG + d] == NL - 1) ? 0.f : sc[d] / den;  // attend mask
      wsh[d] = w;
      sw += w;
    }
    swout[bl] = sw;   // scales attend bias downstream
  }
  __syncthreads();

  float acc = 0.f;
#pragma unroll
  for (int d = 0; d < NDEG; ++d) acc += wsh[d] * nfv[d];
  wsum_bf[(size_t)bl * NFP + t] = (bf16_t)acc;
}

// GRU combine (PyTorch gate order r,z,n): h' = (1-z)*n + z*h.
__global__ void gru_combine_kernel(const float* __restrict__ gi,
                                   const float* __restrict__ gh,
                                   const float* __restrict__ h,
                                   float* __restrict__ hout,
                                   bf16_t* __restrict__ hout_bf)
{
  size_t idx = (size_t)blockIdx.x * blockDim.x + threadIdx.x;
  size_t m = idx >> 8;
  int n = (int)(idx & 255);
  const float* gip = gi + m * (3 * NFP);
  const float* ghp = gh + m * (3 * NFP);
  float r  = sigmoidf_(gip[n] + ghp[n]);
  float z  = sigmoidf_(gip[NFP + n] + ghp[NFP + n]);
  float nn = tanhf(gip[2 * NFP + n] + r * ghp[2 * NFP + n]);
  float hv = h[idx];
  float o = (1.f - z) * nn + z * hv;
  hout[idx] = o;
  hout_bf[idx] = (bf16_t)o;
}

// mol_feature init: sum_l relu(af[b,l,:]) * mask[b,l]
__global__ void mol_init_kernel(const float* __restrict__ af,
                                const float* __restrict__ mask,
                                float* __restrict__ molf,
                                bf16_t* __restrict__ molf_bf)
{
  int b = blockIdx.x, n = threadIdx.x;
  float acc = 0.f;
  for (int l = 0; l < NL; ++l)
    acc += fmaxf(af[((size_t)b * NL + l) * NFP + n], 0.f) * mask[b * NL + l];
  molf[(size_t)b * NFP + n] = acc;
  molf_bf[(size_t)b * NFP + n] = (bf16_t)acc;
}

// Per-molecule attention over atoms: scores, masked softmax over L,
// weighted atom-feature sum -> [B, FP], plus weight-sum for bias scaling.
__global__ __launch_bounds__(256) void mol_align_kernel(
    const float* __restrict__ molf, const float* __restrict__ af,
    const float* __restrict__ mask,
    const float* __restrict__ mw, const float* __restrict__ mb_p,
    bf16_t* __restrict__ wsum_bf, float* __restrict__ swout)
{
  const int b = blockIdx.x, t = threadIdx.x;
  __shared__ float wl[NL];
  float mb = mb_p[0];
  if (t < NL) {
    float dot = 0.f;
    const float* mfp = molf + (size_t)b * NFP;
    const float* afp = af + ((size_t)b * NL + t) * NFP;
    for (int j = 0; j < NFP; ++j) dot += mfp[j] * mw[j] + afp[j] * mw[NFP + j];
    float s = leakyf_(dot + mb);
    if (mask[b * NL + t] == 0.f) s += NEGC;
    wl[t] = s;
  }
  __syncthreads();
  if (t == 0) {
    float mx = -3.4e38f;
    for (int l = 0; l < NL; ++l) mx = fmaxf(mx, wl[l]);
    float den = 0.f;
    for (int l = 0; l < NL; ++l) { wl[l] = __expf(wl[l] - mx); den += wl[l]; }
    float sw = 0.f;
    for (int l = 0; l < NL; ++l) {
      float w = (wl[l] / den) * mask[b * NL + l];
      wl[l] = w;
      sw += w;
    }
    swout[b] = sw;
  }
  __syncthreads();
  float acc = 0.f;
  for (int l = 0; l < NL; ++l)
    acc += wl[l] * af[((size_t)b * NL + l) * NFP + t];
  wsum_bf[(size_t)b * NFP + t] = (bf16_t)acc;
}

// mol_prediction[b] = molf[b] . out_w + out_b
__global__ void final_out_kernel(const float* __restrict__ molf,
                                 const float* __restrict__ ow,
                                 const float* __restrict__ ob,
                                 float* __restrict__ pred)
{
  int b = blockIdx.x, t = threadIdx.x;
  __shared__ float red[256];
  red[t] = molf[(size_t)b * NFP + t] * ow[t];
  __syncthreads();
  for (int s = 128; s > 0; s >>= 1) {
    if (t < s) red[t] += red[t + s];
    __syncthreads();
  }
  if (t == 0) pred[b] = red[0] + ob[0];
}

// ---------------------------------------------------------------------------

extern "C" void kernel_launch(void* const* d_in, const int* in_sizes, int n_in,
                              void* d_out, int out_size, void* d_ws, size_t ws_size,
                              hipStream_t stream)
{
  (void)in_sizes; (void)n_in; (void)out_size; (void)ws_size;

  const float* atom_list  = (const float*)d_in[0];
  const float* bond_list  = (const float*)d_in[1];
  const int*   adeg       = (const int*)d_in[2];
  const int*   bdeg       = (const int*)d_in[3];
  const float* atom_mask  = (const float*)d_in[4];
  const float* atom_fc_w  = (const float*)d_in[5];
  const float* atom_fc_b  = (const float*)d_in[6];
  const float* nb_fc_w    = (const float*)d_in[7];
  const float* nb_fc_b    = (const float*)d_in[8];
  const float* align_w    = (const float*)d_in[9];   // [3, 512]
  const float* align_b    = (const float*)d_in[10];  // [3]
  const float* attend_w   = (const float*)d_in[11];  // [3,256,256]
  const float* attend_b   = (const float*)d_in[12];  // [3,256]
  const float* gru_wih    = (const float*)d_in[13];  // [3,768,256]
  const float* gru_whh    = (const float*)d_in[14];
  const float* gru_bih    = (const float*)d_in[15];  // [3,768]
  const float* gru_bhh    = (const float*)d_in[16];
  const float* mol_align_w  = (const float*)d_in[17]; // [512]
  const float* mol_align_b  = (const float*)d_in[18]; // [1]
  const float* mol_attend_w = (const float*)d_in[19]; // [256,256]
  const float* mol_attend_b = (const float*)d_in[20];
  const float* mol_gru_wih  = (const float*)d_in[21]; // [768,256]
  const float* mol_gru_whh  = (const float*)d_in[22];
  const float* mol_gru_bih  = (const float*)d_in[23];
  const float* mol_gru_bhh  = (const float*)d_in[24];
  const float* out_w = (const float*)d_in[25];
  const float* out_b = (const float*)d_in[26];

  const int ML = NB * NL;         // 16384 atom rows
  const int MN = ML * NDEG;       // 98304 neighbor rows

  // ---- workspace carve (~164 MB; gi aliases consumed nf buffer) ----
  char* p = (char*)d_ws;
  auto take = [&](size_t bytes) { char* q = p; p += (bytes + 255) & ~(size_t)255; return q; };

  bf16_t* ain_bf  = (bf16_t*)take((size_t)ML * 64 * 2);
  bf16_t* nbin_bf = (bf16_t*)take((size_t)MN * 64 * 2);
  char*   nf_gi   = take((size_t)MN * NFP * 2);      // nf_bf, later gi (same size)
  bf16_t* nf_bf   = (bf16_t*)nf_gi;
  float*  gi      = (float*)nf_gi;                   // [ML, 768] fp32
  float*  gh      = (float*)take((size_t)ML * 3 * NFP * 4);
  float*  af      = (float*)take((size_t)ML * NFP * 4);
  bf16_t* af_bf   = (bf16_t*)take((size_t)ML * NFP * 2);
  bf16_t* wsum_bf = (bf16_t*)take((size_t)ML * NFP * 2);
  float*  swv     = (float*)take((size_t)ML * 4);
  bf16_t* ctx_bf  = (bf16_t*)take((size_t)ML * NFP * 2);
  bf16_t* w_atom  = (bf16_t*)take(256 * 64 * 2);
  bf16_t* w_nb    = (bf16_t*)take(256 * 64 * 2);
  bf16_t* w_att   = (bf16_t*)take((size_t)3 * 256 * 256 * 2);
  bf16_t* w_wih   = (bf16_t*)take((size_t)3 * 768 * 256 * 2);
  bf16_t* w_whh   = (bf16_t*)take((size_t)3 * 768 * 256 * 2);
  bf16_t* w_matt  = (bf16_t*)take((size_t)256 * 256 * 2);
  bf16_t* w_mwih  = (bf16_t*)take((size_t)768 * 256 * 2);
  bf16_t* w_mwhh  = (bf16_t*)take((size_t)768 * 256 * 2);
  float*  molf    = (float*)take((size_t)NB * NFP * 4);
  bf16_t* molf_bf = (bf16_t*)take((size_t)NB * NFP * 2);
  bf16_t* mwsum   = (bf16_t*)take((size_t)NB * NFP * 2);
  float*  msw     = (float*)take((size_t)NB * 4);
  bf16_t* mctx_bf = (bf16_t*)take((size_t)NB * NFP * 2);
  float*  mgi     = (float*)take((size_t)NB * 3 * NFP * 4);
  float*  mgh     = (float*)take((size_t)NB * 3 * NFP * 4);

  auto cvt = [&](const float* src, bf16_t* dst, int N, int K, int Kp) {
    size_t total = (size_t)N * Kp;
    convert_pad_kernel<<<(unsigned)((total + 255) / 256), 256, 0, stream>>>(src, dst, K, Kp, total);
  };

  // ---- weight + input conversions (fp32 -> bf16, K padded to 32-mult) ----
  cvt(atom_fc_w, w_atom, 256, NFEAT, 64);
  cvt(nb_fc_w,   w_nb,   256, NFEAT + NBOND, 64);
  cvt(attend_w,  w_att,  3 * 256, 256, 256);
  cvt(gru_wih,   w_wih,  3 * 768, 256, 256);
  cvt(gru_whh,   w_whh,  3 * 768, 256, 256);
  cvt(mol_attend_w, w_matt, 256, 256, 256);
  cvt(mol_gru_wih,  w_mwih, 768, 256, 256);
  cvt(mol_gru_whh,  w_mwhh, 768, 256, 256);
  cvt(atom_list, ain_bf, ML, NFEAT, 64);

  {
    size_t total = (size_t)MN * 64;
    build_neighbor_kernel<<<(unsigned)((total + 255) / 256), 256, 0, stream>>>(
        atom_list, bond_list, adeg, bdeg, nbin_bf);
  }

  // ---- input FCs (WMMA) ----
  gemm_bf16_kernel<1><<<dim3(ML / 128, NFP / 64), 256, 0, stream>>>(
      ain_bf, w_atom, atom_fc_b, nullptr, af, af_bf, ML, NFP, 64);
  gemm_bf16_kernel<1><<<dim3(MN / 128, NFP / 64), 256, 0, stream>>>(
      nbin_bf, w_nb, nb_fc_b, nullptr, nullptr, nf_bf, MN, NFP, 64);

  // ---- radius iterations ----
  for (int d = 0; d < 3; ++d) {
    align_attend_kernel<<<ML, 256, 0, stream>>>(
        af, (d == 0) ? nf_bf : nullptr, adeg,
        align_w + (size_t)d * 512, align_b + d, wsum_bf, swv, d > 0);

    // context = elu(wsum @ attend_w^T + sw * attend_b)
    gemm_bf16_kernel<2><<<dim3(ML / 128, NFP / 64), 256, 0, stream>>>(
        wsum_bf, w_att + (size_t)d * 256 * 256, attend_b + (size_t)d * 256, swv,
        nullptr, ctx_bf, ML, NFP, 256);

    // GRU gates (gi aliases nf_bf region: nf consumed above for d==0)
    gemm_bf16_kernel<0><<<dim3(ML / 128, (3 * NFP) / 64), 256, 0, stream>>>(
        ctx_bf, w_wih + (size_t)d * 768 * 256, gru_bih + (size_t)d * 768, nullptr,
        gi, nullptr, ML, 3 * NFP, 256);
    gemm_bf16_kernel<0><<<dim3(ML / 128, (3 * NFP) / 64), 256, 0, stream>>>(
        af_bf, w_whh + (size_t)d * 768 * 256, gru_bhh + (size_t)d * 768, nullptr,
        gh, nullptr, ML, 3 * NFP, 256);

    gru_combine_kernel<<<ML, 256, 0, stream>>>(gi, gh, af, af, af_bf);
  }

  // ---- molecule phase ----
  mol_init_kernel<<<NB, 256, 0, stream>>>(af, atom_mask, molf, molf_bf);

  for (int t = 0; t < 2; ++t) {
    mol_align_kernel<<<NB, 256, 0, stream>>>(
        molf, af, atom_mask, mol_align_w, mol_align_b, mwsum, msw);
    gemm_bf16_kernel<2><<<dim3(NB / 128, NFP / 64), 256, 0, stream>>>(
        mwsum, w_matt, mol_attend_b, msw, nullptr, mctx_bf, NB, NFP, 256);
    gemm_bf16_kernel<0><<<dim3(NB / 128, (3 * NFP) / 64), 256, 0, stream>>>(
        mctx_bf, w_mwih, mol_gru_bih, nullptr, mgi, nullptr, NB, 3 * NFP, 256);
    gemm_bf16_kernel<0><<<dim3(NB / 128, (3 * NFP) / 64), 256, 0, stream>>>(
        molf_bf, w_mwhh, mol_gru_bhh, nullptr, mgh, nullptr, NB, 3 * NFP, 256);
    gru_combine_kernel<<<NB, 256, 0, stream>>>(mgi, mgh, molf, molf, molf_bf);
  }

  // ---- outputs: atom_feature [B,L,FP] then mol_prediction [B,1] ----
  hipMemcpyAsync(d_out, af, (size_t)ML * NFP * sizeof(float),
                 hipMemcpyDeviceToDevice, stream);
  final_out_kernel<<<NB, 256, 0, stream>>>(
      molf, out_w, out_b, (float*)d_out + (size_t)ML * NFP);
}
